// TransformerEncoderLayer_91096256348736
// MI455X (gfx1250) — matrix operand order repaired and checked by reading, not compiled, not run on hardware
//
#include <hip/hip_runtime.h>

// ---------------------------------------------------------------------------
// Transformer encoder layer for MI455X (gfx1250), bf16 WMMA compute path,
// async global->LDS DMA staging with LDS double buffering.
// B=4, S=1024, D=1024, H=16, Dh=64, FF=4096.
// ---------------------------------------------------------------------------

typedef __attribute__((ext_vector_type(16))) __bf16 v16bf;
typedef __attribute__((ext_vector_type(8)))  float  v8f;
typedef __attribute__((ext_vector_type(8)))  unsigned int v8u;

#define D_MODEL 1024
#define N_HEADS 16
#define D_HEAD  64
#define D_FF    4096
#define BATCH   4
#define SEQ     1024
#define NTOK    (BATCH * SEQ)   // 4096

// f32 -> bf16 round-to-nearest-even
__device__ __forceinline__ unsigned short f2bf(float f) {
    unsigned u = __float_as_uint(f);
    unsigned r = u + 0x7FFFu + ((u >> 16) & 1u);
    return (unsigned short)(r >> 16);
}

// --------------------------- block reductions ------------------------------
__device__ __forceinline__ float blk_sum(float v, float* red, int tid) {
    int lane = tid & 31, wave = tid >> 5;
#pragma unroll
    for (int o = 16; o > 0; o >>= 1) v += __shfl_down(v, o, 32);
    if (lane == 0) red[wave] = v;
    __syncthreads();
    if (wave == 0) {
        float t = (lane < 8) ? red[lane] : 0.0f;
#pragma unroll
        for (int o = 4; o > 0; o >>= 1) t += __shfl_down(t, o, 32);
        if (lane == 0) red[0] = t;
    }
    __syncthreads();
    float r = red[0];
    __syncthreads();
    return r;
}

__device__ __forceinline__ float blk_max(float v, float* red, int tid) {
    int lane = tid & 31, wave = tid >> 5;
#pragma unroll
    for (int o = 16; o > 0; o >>= 1) v = fmaxf(v, __shfl_down(v, o, 32));
    if (lane == 0) red[wave] = v;
    __syncthreads();
    if (wave == 0) {
        float t = (lane < 8) ? red[lane] : -3.402823466e38f;
#pragma unroll
        for (int o = 4; o > 0; o >>= 1) t = fmaxf(t, __shfl_down(t, o, 32));
        if (lane == 0) red[0] = t;
    }
    __syncthreads();
    float r = red[0];
    __syncthreads();
    return r;
}

// --------------------------- small kernels ---------------------------------
__global__ void cvt_bf16_kernel(const float* __restrict__ s,
                                unsigned short* __restrict__ d, int n) {
    int i = blockIdx.x * 256 + threadIdx.x;
    if (i < n) d[i] = f2bf(s[i]);
}

// LayerNorm (unbiased var /(n-1), divide by (std + eps)), bf16 output.
__global__ __launch_bounds__(256) void ln_kernel(const float* __restrict__ x,
                                                 const float* __restrict__ g,
                                                 const float* __restrict__ b,
                                                 unsigned short* __restrict__ out) {
    __shared__ float red[8];
    const int tid = threadIdx.x;
    const long t = blockIdx.x;
    const float4 v = *(const float4*)(x + t * D_MODEL + tid * 4);
    float mean = blk_sum(v.x + v.y + v.z + v.w, red, tid) * (1.0f / D_MODEL);
    float d0 = v.x - mean, d1 = v.y - mean, d2 = v.z - mean, d3 = v.w - mean;
    float ss = blk_sum(d0 * d0 + d1 * d1 + d2 * d2 + d3 * d3, red, tid);
    float inv = 1.0f / (sqrtf(ss * (1.0f / (D_MODEL - 1))) + 1e-12f);
    const float4 gv = *(const float4*)(g + tid * 4);
    const float4 bv = *(const float4*)(b + tid * 4);
    unsigned short* o = out + t * D_MODEL + tid * 4;
    o[0] = f2bf(gv.x * d0 * inv + bv.x);
    o[1] = f2bf(gv.y * d1 * inv + bv.y);
    o[2] = f2bf(gv.z * d2 * inv + bv.z);
    o[3] = f2bf(gv.w * d3 * inv + bv.w);
}

// Row softmax over length-1024 rows, in place (f32).
__global__ __launch_bounds__(256) void softmax_kernel(float* __restrict__ att) {
    __shared__ float red[8];
    const int tid = threadIdx.x;
    float* p = att + (long)blockIdx.x * SEQ + tid * 4;
    float4 v = *(const float4*)p;
    float mx = blk_max(fmaxf(fmaxf(v.x, v.y), fmaxf(v.z, v.w)), red, tid);
    v.x = __expf(v.x - mx); v.y = __expf(v.y - mx);
    v.z = __expf(v.z - mx); v.w = __expf(v.w - mx);
    float s = blk_sum(v.x + v.y + v.z + v.w, red, tid);
    float inv = 1.0f / s;
    v.x *= inv; v.y *= inv; v.z *= inv; v.w *= inv;
    *(float4*)p = v;
}

// --------------------------- WMMA fragments --------------------------------
// LDS tile row stride (elements): BK=32 + 8 pad => 16B-aligned b128 traffic.
#define LDT 40

// A fragment: 16x32 bf16, M = lane&15, K pairs per CDNA5 ISA table.
__device__ __forceinline__ v16bf frag_a(const unsigned short* As, int m0, int lane) {
    const unsigned short* row = As + (m0 + (lane & 15)) * LDT;
    const int kb = (lane & 16) ? 8 : 0;
    v8u u;
#pragma unroll
    for (int r = 0; r < 8; ++r) {
        int k = ((r & 4) << 2) + kb + ((r & 3) << 1);   // {0,16} + kb + 2*(r&3)
        u[r] = *(const unsigned int*)(row + k);
    }
    return __builtin_bit_cast(v16bf, u);
}

// B fragment: 32x16 bf16 stored transposed in LDS as [N][K].
// N = lane&15; K pairs per CDNA5 ISA table (same order as frag_a).
__device__ __forceinline__ v16bf frag_b(const unsigned short* Bs, int n0, int lane) {
    const unsigned short* row = Bs + (n0 + (lane & 15)) * LDT;
    const int kb = (lane & 16) ? 8 : 0;
    v8u u;
#pragma unroll
    for (int r = 0; r < 8; ++r) u[r] = *(const unsigned int*)(row + ((r & 4) << 2) + kb + ((r & 3) << 1));
    return __builtin_bit_cast(v16bf, u);
}

// Async DMA: copy 16B from global to LDS, bypassing VGPRs (ASYNCcnt-tracked).
__device__ __forceinline__ void async_ld_b128(const unsigned short* gptr,
                                              unsigned short* lptr) {
    unsigned lds_addr = (unsigned)(unsigned long long)(uintptr_t)lptr;
    unsigned long long gaddr = (unsigned long long)(uintptr_t)gptr;
    asm volatile("global_load_async_to_lds_b128 %0, %1, off"
                 :: "v"(lds_addr), "v"(gaddr) : "memory");
}
__device__ __forceinline__ void wait_async0() {
    asm volatile("s_wait_asynccnt 0x0" ::: "memory");
}

// --------------------------- generic WMMA GEMM -----------------------------
// C[M,N] = scale * (A[M,K] x B) (+bias[n]) (+res[m,n]) (relu?) ; out f32/bf16.
// B_NK=true : B source is row-major [N,K] (weight-style, already K-transposed)
// B_NK=false: B source is [K,N] with leading dim ldb (e.g. V for attn context)
// A_F32=true: A source is f32, converted to bf16 while staging to LDS.
// When A is bf16 and B is weight-style, uses async global->LDS DMA with LDS
// double-buffering to overlap the next tile's HBM traffic with WMMA compute.
// Batched via blockIdx.z: off = (z/HZ)*S_b + (z%HZ)*S_h for A/B/C/res.
template <int BM, int BN, bool A_F32, bool B_NK, bool RELU, bool BIAS, bool RES, bool OUTBF>
__global__ __launch_bounds__(256, 2) void gemm_wmma_kernel(
    const void* __restrict__ Ap, const unsigned short* __restrict__ Bp,
    const float* __restrict__ bias, const float* __restrict__ resp,
    void* __restrict__ Cp, int K, int lda, int ldb, int ldc, int ldres, int HZ,
    long aSb, long aSh, long bSb, long bSh, long cSb, long cSh, long rSb, long rSh,
    float scale) {
    constexpr int BK = 32, WM = 2, WN = 4;
    constexpr int WTM = BM / WM, WTN = BN / WN;
    constexpr int FM = WTM / 16, FN = WTN / 16;
    constexpr bool USE_ASYNC = (!A_F32) && B_NK;
    constexpr int NBUF = USE_ASYNC ? 2 : 1;
    static_assert(BM % (WM * 16) == 0 && BN % (WN * 16) == 0, "tile cfg");

    __shared__ alignas(16) unsigned short As[NBUF * BM * LDT];
    __shared__ alignas(16) unsigned short Bs[NBUF * BN * LDT];

    const int tid = threadIdx.x, lane = tid & 31, wave = tid >> 5;
    const int wm = wave % WM, wn = wave / WM;
    const int zb = (int)blockIdx.z / HZ, zh = (int)blockIdx.z % HZ;
    const long aOff = (long)zb * aSb + (long)zh * aSh;
    const long bOff = (long)zb * bSb + (long)zh * bSh;
    const long cOff = (long)zb * cSb + (long)zh * cSh;
    const long rOff = (long)zb * rSb + (long)zh * rSh;
    const long rowBase = (long)blockIdx.y * BM;
    const long colBase = (long)blockIdx.x * BN;

    const unsigned short* Ab = nullptr;
    const float* Af = nullptr;
    if constexpr (A_F32) Af = (const float*)Ap + aOff;
    else                 Ab = (const unsigned short*)Ap + aOff;
    const unsigned short* Bb = Bp + bOff;

    v8f acc[FM][FN] = {};

    if constexpr (USE_ASYNC) {
        // Each thread owns 2 A-slots + 2 B-slots of 8 bf16 (b128) per tile.
        constexpr int ASL = BM * BK / 8 / 256;  // 2 for BM=128
        constexpr int BSL = BN * BK / 8 / 256;  // 2 for BN=128
        auto issue_tile = [&](int k0, int buf) {
            unsigned short* Ad = As + buf * (BM * LDT);
            unsigned short* Bd = Bs + buf * (BN * LDT);
#pragma unroll
            for (int s = 0; s < ASL; ++s) {
                int i = tid + s * 256, row = i >> 2, c = i & 3;
                async_ld_b128(Ab + (rowBase + row) * lda + k0 + c * 8,
                              Ad + row * LDT + c * 8);
            }
#pragma unroll
            for (int s = 0; s < BSL; ++s) {
                int i = tid + s * 256, row = i >> 2, c = i & 3;
                async_ld_b128(Bb + (colBase + row) * ldb + k0 + c * 8,
                              Bd + row * LDT + c * 8);
            }
        };
        const int nT = K / BK;
        issue_tile(0, 0);
        for (int t = 0; t < nT; ++t) {
            wait_async0();        // tile t landed in LDS (this wave's DMAs)
            __syncthreads();      // ... and everyone else's; prev reads done
            if (t + 1 < nT) issue_tile((t + 1) * BK, (t + 1) & 1);
            const unsigned short* Ac = As + (t & 1) * (BM * LDT);
            const unsigned short* Bc = Bs + (t & 1) * (BN * LDT);
            v16bf af[FM], bfv[FN];
#pragma unroll
            for (int mi = 0; mi < FM; ++mi) af[mi] = frag_a(Ac, wm * WTM + mi * 16, lane);
#pragma unroll
            for (int nj = 0; nj < FN; ++nj) bfv[nj] = frag_b(Bc, wn * WTN + nj * 16, lane);
#pragma unroll
            for (int mi = 0; mi < FM; ++mi)
#pragma unroll
                for (int nj = 0; nj < FN; ++nj)
                    acc[mi][nj] = __builtin_amdgcn_wmma_f32_16x16x32_bf16(
                        false, af[mi], false, bfv[nj], (short)0, acc[mi][nj], false, false);
        }
    } else {
        for (int k0 = 0; k0 < K; k0 += BK) {
            // ---- stage A tile [BM][BK] into LDS (f32 -> bf16 convert) ----
            if constexpr (A_F32) {
#pragma unroll
                for (int s = 0; s < BM * BK / 4 / 256; ++s) {
                    int i = tid + s * 256, row = i >> 3, c = i & 7;
                    float4 f = *(const float4*)(Af + (rowBase + row) * lda + k0 + c * 4);
                    unsigned short* dst = &As[row * LDT + c * 4];
                    dst[0] = f2bf(f.x); dst[1] = f2bf(f.y);
                    dst[2] = f2bf(f.z); dst[3] = f2bf(f.w);
                }
            } else {
#pragma unroll
                for (int s = 0; s < BM * BK / 8 / 256; ++s) {
                    int i = tid + s * 256, row = i >> 2, c = i & 3;
                    *(uint4*)&As[row * LDT + c * 8] =
                        *(const uint4*)(Ab + (rowBase + row) * lda + k0 + c * 8);
                }
            }
            // ---- stage B tile into LDS as [BN][BK] (K-transposed) ----
            if constexpr (B_NK) {
#pragma unroll
                for (int s = 0; s < BN * BK / 8 / 256; ++s) {
                    int i = tid + s * 256, row = i >> 2, c = i & 3;
                    *(uint4*)&Bs[row * LDT + c * 8] =
                        *(const uint4*)(Bb + (colBase + row) * ldb + k0 + c * 8);
                }
            } else {
#pragma unroll
                for (int s = 0; s < BK * BN / 8 / 256; ++s) {
                    int i = tid + s * 256;
                    int k = i / (BN / 8), c = i % (BN / 8);
                    uint4 d = *(const uint4*)(Bb + (long)(k0 + k) * ldb + colBase + c * 8);
                    unsigned v[4] = {d.x, d.y, d.z, d.w};
#pragma unroll
                    for (int j = 0; j < 4; ++j) {
                        Bs[(c * 8 + 2 * j + 0) * LDT + k] = (unsigned short)(v[j] & 0xffffu);
                        Bs[(c * 8 + 2 * j + 1) * LDT + k] = (unsigned short)(v[j] >> 16);
                    }
                }
            }
            __syncthreads();
            v16bf af[FM], bfv[FN];
#pragma unroll
            for (int mi = 0; mi < FM; ++mi) af[mi] = frag_a(As, wm * WTM + mi * 16, lane);
#pragma unroll
            for (int nj = 0; nj < FN; ++nj) bfv[nj] = frag_b(Bs, wn * WTN + nj * 16, lane);
#pragma unroll
            for (int mi = 0; mi < FM; ++mi)
#pragma unroll
                for (int nj = 0; nj < FN; ++nj)
                    acc[mi][nj] = __builtin_amdgcn_wmma_f32_16x16x32_bf16(
                        false, af[mi], false, bfv[nj], (short)0, acc[mi][nj], false, false);
            __syncthreads();
        }
    }

    // ---- epilogue ----
    float* Cf = nullptr;
    unsigned short* Cbf = nullptr;
    if constexpr (OUTBF) Cbf = (unsigned short*)Cp + cOff;
    else                 Cf = (float*)Cp + cOff;
    const float* Rp = nullptr;
    if constexpr (RES) Rp = resp + rOff;

#pragma unroll
    for (int mi = 0; mi < FM; ++mi) {
#pragma unroll
        for (int nj = 0; nj < FN; ++nj) {
            const long n = colBase + wn * WTN + nj * 16 + (lane & 15);
            const long mBase = rowBase + wm * WTM + mi * 16 + ((lane & 16) ? 8 : 0);
            float bv = 0.0f;
            if constexpr (BIAS) bv = bias[n];
#pragma unroll
            for (int r = 0; r < 8; ++r) {
                long m = mBase + r;
                float val = acc[mi][nj][r] * scale + bv;
                if constexpr (RES) val += Rp[m * ldres + n];
                if constexpr (RELU) val = val > 0.0f ? val : 0.0f;
                if constexpr (OUTBF) Cbf[m * ldc + n] = f2bf(val);
                else                 Cf[m * ldc + n] = val;
            }
        }
    }
}

// --------------------------- host launcher ---------------------------------
extern "C" void kernel_launch(void* const* d_in, const int* in_sizes, int n_in,
                              void* d_out, int out_size, void* d_ws, size_t ws_size,
                              hipStream_t stream) {
    (void)in_sizes; (void)n_in; (void)out_size; (void)ws_size;
    const float* x     = (const float*)d_in[0];
    const float* ln1_g = (const float*)d_in[1];
    const float* ln1_b = (const float*)d_in[2];
    const float* wq    = (const float*)d_in[3];
    const float* bq    = (const float*)d_in[4];
    const float* wk    = (const float*)d_in[5];
    const float* bk    = (const float*)d_in[6];
    const float* wv    = (const float*)d_in[7];
    const float* bv    = (const float*)d_in[8];
    const float* ln2_g = (const float*)d_in[9];
    const float* ln2_b = (const float*)d_in[10];
    const float* w1    = (const float*)d_in[11];
    const float* b1    = (const float*)d_in[12];
    const float* w2    = (const float*)d_in[13];
    const float* b2    = (const float*)d_in[14];

    // workspace carve-up (256B aligned)
    char* wp = (char*)d_ws;
    auto carve = [&](size_t bytes) {
        char* p = wp;
        wp += (bytes + 255) & ~(size_t)255;
        return p;
    };
    unsigned short* hb  = (unsigned short*)carve((size_t)NTOK * D_MODEL * 2);
    unsigned short* qb  = (unsigned short*)carve((size_t)NTOK * D_MODEL * 2);
    unsigned short* kb_ = (unsigned short*)carve((size_t)NTOK * D_MODEL * 2);
    unsigned short* vb_ = (unsigned short*)carve((size_t)NTOK * D_MODEL * 2);
    float*          o1  = (float*)carve((size_t)NTOK * D_MODEL * 4);
    unsigned short* h2b = (unsigned short*)carve((size_t)NTOK * D_MODEL * 2);
    unsigned short* fb  = (unsigned short*)carve((size_t)NTOK * D_FF * 2);
    unsigned short* wqb = (unsigned short*)carve((size_t)D_MODEL * D_MODEL * 2);
    unsigned short* wkb = (unsigned short*)carve((size_t)D_MODEL * D_MODEL * 2);
    unsigned short* wvb = (unsigned short*)carve((size_t)D_MODEL * D_MODEL * 2);
    unsigned short* w1b = (unsigned short*)carve((size_t)D_FF * D_MODEL * 2);
    unsigned short* w2b = (unsigned short*)carve((size_t)D_MODEL * D_FF * 2);

    float* att = (float*)d_out + (size_t)NTOK * D_MODEL;  // [H*B, S, S]

    const int nW = D_MODEL * D_MODEL;   // 1M
    const int nF = D_FF * D_MODEL;      // 4M
    cvt_bf16_kernel<<<(nW + 255) / 256, 256, 0, stream>>>(wq, wqb, nW);
    cvt_bf16_kernel<<<(nW + 255) / 256, 256, 0, stream>>>(wk, wkb, nW);
    cvt_bf16_kernel<<<(nW + 255) / 256, 256, 0, stream>>>(wv, wvb, nW);
    cvt_bf16_kernel<<<(nF + 255) / 256, 256, 0, stream>>>(w1, w1b, nF);
    cvt_bf16_kernel<<<(nF + 255) / 256, 256, 0, stream>>>(w2, w2b, nF);

    // LN1 -> hb (bf16)
    ln_kernel<<<NTOK, 256, 0, stream>>>(x, ln1_g, ln1_b, hb);

    // QKV projections: [4096,1024] = hb @ W^T + b, bf16 out
    dim3 gQKV(D_MODEL / 128, NTOK / 128, 1);
    gemm_wmma_kernel<128, 128, false, true, false, true, false, true>
        <<<gQKV, 256, 0, stream>>>(hb, wqb, bq, nullptr, qb,
            D_MODEL, D_MODEL, D_MODEL, D_MODEL, 0, 1, 0,0,0,0,0,0,0,0, 1.0f);
    gemm_wmma_kernel<128, 128, false, true, false, true, false, true>
        <<<gQKV, 256, 0, stream>>>(hb, wkb, bk, nullptr, kb_,
            D_MODEL, D_MODEL, D_MODEL, D_MODEL, 0, 1, 0,0,0,0,0,0,0,0, 1.0f);
    gemm_wmma_kernel<128, 128, false, true, false, true, false, true>
        <<<gQKV, 256, 0, stream>>>(hb, wvb, bv, nullptr, vb_,
            D_MODEL, D_MODEL, D_MODEL, D_MODEL, 0, 1, 0,0,0,0,0,0,0,0, 1.0f);

    // scores[h*B+b, q, k] = (Q . K^T) / 8  -> f32 straight into d_out attn
    dim3 gSc(SEQ / 128, SEQ / 128, BATCH * N_HEADS);
    gemm_wmma_kernel<128, 128, false, true, false, false, false, false>
        <<<gSc, 256, 0, stream>>>(qb, kb_, nullptr, nullptr, att,
            D_HEAD, D_MODEL, D_MODEL, SEQ, 0, N_HEADS,
            (long)SEQ * D_MODEL, D_HEAD,               // A: b*S*D + h*64
            (long)SEQ * D_MODEL, D_HEAD,               // B: b*S*D + h*64
            (long)SEQ * SEQ, (long)BATCH * SEQ * SEQ,  // C: (h*B+b)*S*S
            0, 0, 0.125f);

    // softmax rows, in place
    softmax_kernel<<<N_HEADS * BATCH * SEQ, 256, 0, stream>>>(att);

    // ctx + residual(x): o1[b*S+q, h*64+d] = attn @ V + x
    dim3 gCtx(D_HEAD / 64, SEQ / 128, BATCH * N_HEADS);
    gemm_wmma_kernel<128, 64, true, false, false, false, true, false>
        <<<gCtx, 256, 0, stream>>>(att, vb_, nullptr, x, o1,
            SEQ, SEQ, D_MODEL, D_MODEL, D_MODEL, N_HEADS,
            (long)SEQ * SEQ, (long)BATCH * SEQ * SEQ,  // A: attn
            (long)SEQ * D_MODEL, D_HEAD,               // B: V slice [K=S rows]
            (long)SEQ * D_MODEL, D_HEAD,               // C: o1 slice
            (long)SEQ * D_MODEL, D_HEAD,               // res: x slice
            1.0f);

    // LN2 -> h2b (bf16)
    ln_kernel<<<NTOK, 256, 0, stream>>>(o1, ln2_g, ln2_b, h2b);

    // FFN1: fb = relu(h2 @ w1^T + b1), bf16 out [4096, 4096]
    dim3 gF1(D_FF / 128, NTOK / 128, 1);
    gemm_wmma_kernel<128, 128, false, true, true, true, false, true>
        <<<gF1, 256, 0, stream>>>(h2b, w1b, b1, nullptr, fb,
            D_MODEL, D_MODEL, D_MODEL, D_FF, 0, 1, 0,0,0,0,0,0,0,0, 1.0f);

    // FFN2: out = fb @ w2^T + b2 + o1, f32 out -> d_out[0 .. T*D)
    dim3 gF2(D_MODEL / 128, NTOK / 128, 1);
    gemm_wmma_kernel<128, 128, false, true, false, true, true, false>
        <<<gF2, 256, 0, stream>>>(fb, w2b, b2, o1, d_out,
            D_FF, D_FF, D_FF, D_MODEL, D_MODEL, 1, 0,0,0,0,0,0,0,0, 1.0f);
}
